// HierarchicalGRU_54374285968083
// MI455X (gfx1250) — matrix operand order repaired and checked by reading, not compiled, not run on hardware
//
#include <hip/hip_runtime.h>
#include <hip/hip_bf16.h>
#include <stdint.h>

// ---------------- problem constants ----------------
#define B_   16
#define U_   64
#define WT_  64
#define E_   256
#define H2_  128
#define H_   256
#define NSEQ (B_ * U_)        // 1024 word sequences
#define MW   (NSEQ * WT_)     // 65536 rows of word-level input-projection GEMM
#define MU   (B_ * U_)        // 1024 rows of utterance-level input-projection GEMM
#define G3   (3 * H2_)        // 384 gate outputs per direction

typedef __attribute__((ext_vector_type(16))) __bf16 v16bf;
typedef __attribute__((ext_vector_type(8)))  float  v8f;
typedef unsigned int u32x4 __attribute__((ext_vector_type(4)));
typedef int          i32x4 __attribute__((ext_vector_type(4)));
typedef int          i32x8 __attribute__((ext_vector_type(8)));

// ---------------- bf16 helpers (RNE) ----------------
__device__ __forceinline__ unsigned short f2bf_bits(float f) {
  unsigned u = __float_as_uint(f);
  unsigned r = (u + 0x7FFFu + ((u >> 16) & 1u)) >> 16;
  return (unsigned short)r;
}
__device__ __forceinline__ unsigned pack2bf(float a, float b) {
  return (unsigned)f2bf_bits(a) | ((unsigned)f2bf_bits(b) << 16);
}

// ---------------- WMMA fragment loaders (ISA 7.12.2 layouts) ----------------
union FragU { v16bf v; uint4 q[2]; };

// 16-bit A 16x32: lanes 0-15 -> row M=lane, K{0..7} then K{16..23};
//                 lanes 16-31 -> row M=lane-16, K{8..15} then K{24..31}.
__device__ __forceinline__ v16bf load_a_frag(const unsigned short* base, int stride) {
  int lane = threadIdx.x & 31;
  const unsigned short* p = base + (lane & 15) * stride + ((lane >> 4) << 3);
  FragU u;
  u.q[0] = *reinterpret_cast<const uint4*>(p);
  u.q[1] = *reinterpret_cast<const uint4*>(p + 16);
  return u.v;
}
// 16-bit B 32x16: col N = lane&15; lanes 0-15 hold K{0..7},K{16..23}, lanes 16-31 K{8..15},K{24..31}.
// Storage is B[n][k] contiguous in k (i.e. W[g][k]), so each lane reads two runs of 8 contiguous shorts (k and k+16).
__device__ __forceinline__ v16bf load_b_frag(const unsigned short* base, int stride) {
  int lane = threadIdx.x & 31;
  const unsigned short* p = base + (lane & 15) * stride + ((lane >> 4) << 3);
  FragU u;
  u.q[0] = *reinterpret_cast<const uint4*>(p);
  u.q[1] = *reinterpret_cast<const uint4*>(p + 16);
  return u.v;
}
// 32-bit C/D 16x16: VGPR v holds M=v (lanes<16) / M=v+8 (lanes>=16), N=lane&15.
__device__ __forceinline__ void store_c(float* base, int stride, v8f c) {
  int lane = threadIdx.x & 31;
  int n  = lane & 15;
  int m0 = (lane >> 4) << 3;
#pragma unroll
  for (int i = 0; i < 8; ++i) base[(m0 + i) * stride + n] = c[i];
}

__device__ __forceinline__ v8f wmma_bf16(v16bf a, v16bf b, v8f c) {
  return __builtin_amdgcn_wmma_f32_16x16x32_bf16(false, a, false, b, (short)0, c, false, false);
}

// ---------------- TDM: 1D (rows=1) tile of 2-byte elements, global -> LDS ----------------
// D# per CDNA5 ISA: group0 {count=1, lds_addr, global_addr, type=2},
// group1 {data_size=2B, tensor_dim0=tile_dim0=nelem, tensor_dim1=tile_dim1=1, stride0=nelem}.
__device__ __forceinline__ void tdm_load_lds_u16(unsigned lds_off, const void* gptr,
                                                 unsigned nelem) {
  unsigned long long ga = (unsigned long long)(uintptr_t)gptr;
  u32x4 g0;
  g0.x = 1u;                                         // count=1, user descriptor
  g0.y = lds_off;                                    // lds_addr (bytes)
  g0.z = (unsigned)(ga & 0xFFFFFFFFu);               // global_addr[31:0]
  g0.w = (unsigned)((ga >> 32) & 0x1FFFFFFu)         // global_addr[56:32]
       | 0x80000000u;                                // type=2 ("image")
  i32x8 g1;
  g1[0] = 1 << 16;                                   // data_size=1 (2 bytes)
  g1[1] = (int)((nelem & 0xFFFFu) << 16);            // tensor_dim0[15:0] @ bit48
  g1[2] = (int)(((nelem >> 16) & 0xFFFFu) | (1u << 16)); // tensor_dim0[31:16], tensor_dim1=1
  g1[3] = (int)((nelem & 0xFFFFu) << 16);            // tile_dim0 @ bit112
  g1[4] = 1;                                         // tile_dim1=1, tile_dim2=0
  g1[5] = (int)nelem;                                // tensor_dim0_stride[31:0]
  g1[6] = 0;                                         // stride0[47:32], stride1[15:0]
  g1[7] = 0;
  i32x4 z4 = {0, 0, 0, 0};
#if __clang_major__ >= 23
  i32x8 z8 = {0, 0, 0, 0, 0, 0, 0, 0};
  __builtin_amdgcn_tensor_load_to_lds(g0, g1, z4, z4, z8, 0);
#else
  __builtin_amdgcn_tensor_load_to_lds(g0, g1, z4, z4, 0);
#endif
}

// ---------------- kernel 1: fp32 -> bf16 weight concat (fwd rows then bwd rows) ----------------
__global__ void cvt_concat(const float* __restrict__ A, const float* __restrict__ Bm,
                           unsigned short* __restrict__ dst, int half) {
  int i = blockIdx.x * blockDim.x + threadIdx.x;
  if (i >= half) return;
  dst[i]        = f2bf_bits(A[i]);
  dst[half + i] = f2bf_bits(Bm[i]);
}

// ---------------- kernel 2: embedding gather -> bf16 activations ----------------
__global__ void embed_gather(const int* __restrict__ tok, const float* __restrict__ emb,
                             unsigned short* __restrict__ x) {
  int p = blockIdx.x * blockDim.x + threadIdx.x;       // pair index
  if (p >= MW * (E_ / 2)) return;
  int row = p >> 7;
  int e2  = (p & 127) * 2;
  int t   = tok[row];
  const float* src = emb + (size_t)t * E_ + e2;
  float a = (t == 0) ? 0.0f : src[0];
  float b = (t == 0) ? 0.0f : src[1];
  reinterpret_cast<unsigned*>(x)[p] = pack2bf(a, b);
}

// ---------------- kernel 3: bf16 WMMA GEMM  xg[dir][m][g] = x[m]·W[g]ᵀ + bih ----------------
// A: M x 256 bf16 row-major.  Wc: 768 x 256 bf16 (row g = output column, fwd 0..383, bwd 384..767).
// Block tile: 128(M) x 64(N), K-steps of 32. 8 waves, each owns a 16-row strip x 4 N-tiles.
__global__ __launch_bounds__(256) void gemm_xg(
    const unsigned short* __restrict__ A, const unsigned short* __restrict__ Wc,
    const float* __restrict__ bias_f, const float* __restrict__ bias_b,
    float* __restrict__ xg, int M) {
  __shared__ alignas(16) unsigned short sA[128 * 32];
  __shared__ alignas(16) unsigned short sB[64 * 32];
  const int m0 = blockIdx.x * 128;
  const int n0 = blockIdx.y * 64;
  const int tid = threadIdx.x, lane = tid & 31, wave = tid >> 5;

  v8f acc[4] = { {0,0,0,0,0,0,0,0}, {0,0,0,0,0,0,0,0},
                 {0,0,0,0,0,0,0,0}, {0,0,0,0,0,0,0,0} };

  for (int k0 = 0; k0 < E_; k0 += 32) {
    {
      int r = tid >> 1, c = (tid & 1) * 16;
      const unsigned short* g = A + (size_t)(m0 + r) * E_ + k0 + c;
      *reinterpret_cast<uint4*>(&sA[r * 32 + c]) = *reinterpret_cast<const uint4*>(g);
      if (k0 + 32 < E_) __builtin_prefetch(g + 32, 0, 3);   // global_prefetch_b8 (near)
    }
    if (tid < 128) {
      int n = tid >> 1, c = (tid & 1) * 16;
      const unsigned short* g = Wc + (size_t)(n0 + n) * E_ + k0 + c;
      *reinterpret_cast<uint4*>(&sB[n * 32 + c]) = *reinterpret_cast<const uint4*>(g);
    }
    __syncthreads();
    v16bf a = load_a_frag(&sA[(wave * 16) * 32], 32);
#pragma unroll
    for (int nt = 0; nt < 4; ++nt) {
      v16bf b = load_b_frag(&sB[(nt * 16) * 32], 32);
      acc[nt] = wmma_bf16(a, b, acc[nt]);
    }
    __syncthreads();
  }
  // epilogue: + bih, scatter to [dir][m][384]
  const int n_lane = lane & 15;
  const int m_base = m0 + wave * 16 + ((lane >> 4) << 3);
#pragma unroll
  for (int nt = 0; nt < 4; ++nt) {
    int g_total = n0 + nt * 16 + n_lane;
    int dir = (g_total >= G3) ? 1 : 0;
    int g = g_total - dir * G3;
    float bias = dir ? bias_b[g] : bias_f[g];
    float* out = xg + (size_t)dir * M * G3;
#pragma unroll
    for (int i = 0; i < 8; ++i)
      out[(size_t)(m_base + i) * G3 + g] = acc[nt][i] + bias;
  }
}

// ---------------- kernel 4: persistent GRU scan (one block = 16 rows, one direction) ----------------
// Dynamic LDS: Whh bf16 (96KB, TDM-staged) + hg f32 (24KB) + h bf16 (4KB) + h f32 (8KB) = 132KB.
// Per step: 8 waves x 3 N-tiles x 4 K-steps = 96 v_wmma, then VALU gates.
__global__ __launch_bounds__(256) void gru_scan(
    const float* __restrict__ xg,            // [2][NR][T][384]
    const unsigned short* __restrict__ Ucat, // [768][128] bf16 (fwd 0..383, bwd 384..767)
    const float* __restrict__ bh_f, const float* __restrict__ bh_b,
    float* __restrict__ y,                   // [NR][T][256]; dir d -> channels d*128..d*128+127
    int NR, int T) {
  extern __shared__ char smem[];
  unsigned short* sU  = reinterpret_cast<unsigned short*>(smem);               // 384*128
  float*          sHG = reinterpret_cast<float*>(smem + G3 * H2_ * 2);         // 16*384
  unsigned short* sH  = reinterpret_cast<unsigned short*>(smem + G3 * H2_ * 2 + 16 * G3 * 4);
  float*          sHf = reinterpret_cast<float*>(smem + G3 * H2_ * 2 + 16 * G3 * 4 + 16 * H2_ * 2);

  const int dir  = blockIdx.y;
  const int row0 = blockIdx.x * 16;
  const int tid  = threadIdx.x, wave = tid >> 5;
  const float* bh = dir ? bh_b : bh_f;

  // Stage recurrent weights once via the Tensor Data Mover (async DMA, TENSORcnt).
  // One wave issues the descriptor (TDM ignores EXEC; one wave => one DMA), waits
  // TENSORcnt==0, then the workgroup barrier publishes the LDS tile to all waves.
  if (tid < 32) {
    tdm_load_lds_u16((unsigned)(uintptr_t)sU,
                     Ucat + (size_t)dir * G3 * H2_, G3 * H2_);
    __builtin_amdgcn_s_wait_tensorcnt(0);
  }
  for (int i = tid; i < 16 * H2_; i += 256) { sH[i] = 0; sHf[i] = 0.0f; }
  __syncthreads();

  const size_t xg_dir = (size_t)dir * NR * T * G3;

  for (int step = 0; step < T; ++step) {
    const int t = dir ? (T - 1 - step) : step;

    // hg(16x384) = h_bf16(16x128) @ Whhᵀ(128x384)
    v8f a0 = {0,0,0,0,0,0,0,0}, a1 = {0,0,0,0,0,0,0,0}, a2 = {0,0,0,0,0,0,0,0};
#pragma unroll
    for (int kk = 0; kk < 4; ++kk) {
      v16bf af = load_a_frag(&sH[kk * 32], H2_);
      v16bf b0 = load_b_frag(&sU[(wave * 3 + 0) * 16 * H2_ + kk * 32], H2_);
      a0 = wmma_bf16(af, b0, a0);
      v16bf b1 = load_b_frag(&sU[(wave * 3 + 1) * 16 * H2_ + kk * 32], H2_);
      a1 = wmma_bf16(af, b1, a1);
      v16bf b2 = load_b_frag(&sU[(wave * 3 + 2) * 16 * H2_ + kk * 32], H2_);
      a2 = wmma_bf16(af, b2, a2);
    }
    store_c(&sHG[(wave * 3 + 0) * 16], G3, a0);
    store_c(&sHG[(wave * 3 + 1) * 16], G3, a1);
    store_c(&sHG[(wave * 3 + 2) * 16], G3, a2);
    __syncthreads();

    // gates + state update (f32)
#pragma unroll
    for (int q = tid; q < 16 * H2_; q += 256) {
      int m = q >> 7, j = q & (H2_ - 1);
      const float* xp = xg + xg_dir + ((size_t)(row0 + m) * T + t) * G3;
      float xr = xp[j], xz = xp[H2_ + j], xn = xp[2 * H2_ + j];
      float hr = sHG[m * G3 + j]            + bh[j];
      float hz = sHG[m * G3 + H2_ + j]      + bh[H2_ + j];
      float hn = sHG[m * G3 + 2 * H2_ + j]  + bh[2 * H2_ + j];
      float r = 1.0f / (1.0f + __expf(-(xr + hr)));
      float z = 1.0f / (1.0f + __expf(-(xz + hz)));
      float n = tanhf(xn + r * hn);
      float hnew = (1.0f - z) * n + z * sHf[q];
      sHf[q] = hnew;
      sH[q]  = f2bf_bits(hnew);
      y[((size_t)(row0 + m) * T + t) * H_ + dir * H2_ + j] = hnew;
    }
    __syncthreads();
  }
}

// ---------------- kernel 5: pick last valid word step -> bf16 utterance input ----------------
__global__ void select_last(const float* __restrict__ w_out, const int* __restrict__ w_len,
                            unsigned short* __restrict__ utt_in) {
  int row = blockIdx.x;                 // b*64+u
  int tid = threadIdx.x;                // 128 threads, 2 channels each
  int t = w_len[row] - 1;
  const float* src = w_out + ((size_t)row * WT_ + t) * H_;
  reinterpret_cast<unsigned*>(utt_in)[row * (H_ / 2) + tid] =
      pack2bf(src[2 * tid], src[2 * tid + 1]);
}

// ---------------- kernel 6: packing offsets (exclusive cumsum of masked lengths) ----------------
__global__ void compute_offs(const int* __restrict__ u_len, const int* __restrict__ w_len,
                             int* __restrict__ offs, int* __restrict__ wl2) {
  int b = blockIdx.x;
  if (threadIdx.x == 0) {
    int ul = u_len[b], run = 0;
    for (int u = 0; u < U_; ++u) {
      int wl = (u < ul) ? w_len[b * U_ + u] : 0;
      offs[b * U_ + u] = run;
      wl2[b * U_ + u]  = wl;
      run += wl;
    }
  }
}

// ---------------- kernel 7: zero + packed scatter of w_out into w2 ----------------
__global__ void zero_f4(float* __restrict__ p, size_t n) {
  size_t i = ((size_t)blockIdx.x * blockDim.x + threadIdx.x) * 4;
  if (i < n) *reinterpret_cast<float4*>(p + i) = make_float4(0.f, 0.f, 0.f, 0.f);
}
__global__ void scatter_w2(const float* __restrict__ w_out, const int* __restrict__ offs,
                           const int* __restrict__ wl2, float* __restrict__ w2) {
  int bu = blockIdx.x;
  int b  = bu >> 6;
  int wl = wl2[bu], o = offs[bu];
  const float* src = w_out + (size_t)bu * WT_ * H_;
  float* dst = w2 + ((size_t)b * (U_ * WT_) + o) * H_;
  for (int t = 0; t < wl; ++t)
    for (int e = threadIdx.x; e < H_; e += blockDim.x)
      dst[(size_t)t * H_ + e] = src[(size_t)t * H_ + e];
}

// ---------------- orchestration ----------------
extern "C" void kernel_launch(void* const* d_in, const int* in_sizes, int n_in,
                              void* d_out, int out_size, void* d_ws, size_t ws_size,
                              hipStream_t stream) {
  (void)in_sizes; (void)n_in; (void)out_size; (void)ws_size;
  const int*   tok   = (const int*)d_in[0];
  const int*   u_len = (const int*)d_in[1];
  const int*   w_len = (const int*)d_in[2];
  const float* emb   = (const float*)d_in[3];
  const float* Wf_w  = (const float*)d_in[4];
  const float* Uf_w  = (const float*)d_in[5];
  const float* bif_w = (const float*)d_in[6];
  const float* bhf_w = (const float*)d_in[7];
  const float* Wb_w  = (const float*)d_in[8];
  const float* Ub_w  = (const float*)d_in[9];
  const float* bib_w = (const float*)d_in[10];
  const float* bhb_w = (const float*)d_in[11];
  const float* Wf_u  = (const float*)d_in[12];
  const float* Uf_u  = (const float*)d_in[13];
  const float* bif_u = (const float*)d_in[14];
  const float* bhf_u = (const float*)d_in[15];
  const float* Wb_u  = (const float*)d_in[16];
  const float* Ub_u  = (const float*)d_in[17];
  const float* bib_u = (const float*)d_in[18];
  const float* bhb_u = (const float*)d_in[19];

  char* ws = (char*)d_ws;
  size_t off = 0;
  auto carve = [&](size_t bytes) {
    char* p = ws + off; off += (bytes + 255) & ~(size_t)255; return p;
  };
  unsigned short* x_bf   = (unsigned short*)carve((size_t)MW * E_ * 2);
  unsigned short* Wc_w   = (unsigned short*)carve((size_t)2 * G3 * E_ * 2);
  unsigned short* Uc_w   = (unsigned short*)carve((size_t)2 * G3 * H2_ * 2);
  unsigned short* Wc_u   = (unsigned short*)carve((size_t)2 * G3 * E_ * 2);
  unsigned short* Uc_u   = (unsigned short*)carve((size_t)2 * G3 * H2_ * 2);
  float*          xg_w   = (float*)carve((size_t)2 * MW * G3 * 4);
  float*          w_out  = (float*)carve((size_t)MW * H_ * 4);
  unsigned short* utt_in = (unsigned short*)carve((size_t)MU * H_ * 2);
  float*          xg_u   = (float*)carve((size_t)2 * MU * G3 * 4);
  int*            offs   = (int*)carve(NSEQ * 4);
  int*            wl2    = (int*)carve(NSEQ * 4);

  float* u_out = (float*)d_out;                              // (B,U,256)
  float* w2    = (float*)d_out + (size_t)B_ * U_ * H_;       // (B,U*WT,256)

  const int SCAN_LDS = G3 * H2_ * 2 + 16 * G3 * 4 + 16 * H2_ * 2 + 16 * H2_ * 4; // 132KB

  // weights -> bf16
  cvt_concat<<<(G3 * E_  + 255) / 256, 256, 0, stream>>>(Wf_w, Wb_w, Wc_w, G3 * E_);
  cvt_concat<<<(G3 * H2_ + 255) / 256, 256, 0, stream>>>(Uf_w, Ub_w, Uc_w, G3 * H2_);
  cvt_concat<<<(G3 * E_  + 255) / 256, 256, 0, stream>>>(Wf_u, Wb_u, Wc_u, G3 * E_);
  cvt_concat<<<(G3 * H2_ + 255) / 256, 256, 0, stream>>>(Uf_u, Ub_u, Uc_u, G3 * H2_);

  // embedding gather -> bf16 activations
  embed_gather<<<(MW * (E_ / 2)) / 256, 256, 0, stream>>>(tok, emb, x_bf);

  // word-level: input projection GEMM + recurrent scan
  gemm_xg<<<dim3(MW / 128, (2 * G3) / 64), 256, 0, stream>>>(x_bf, Wc_w, bif_w, bib_w, xg_w, MW);
  gru_scan<<<dim3(NSEQ / 16, 2), 256, SCAN_LDS, stream>>>(xg_w, Uc_w, bhf_w, bhb_w, w_out, NSEQ, WT_);

  // utterance-level
  select_last<<<NSEQ, 128, 0, stream>>>(w_out, w_len, utt_in);
  gemm_xg<<<dim3(MU / 128, (2 * G3) / 64), 256, 0, stream>>>(utt_in, Wc_u, bif_u, bib_u, xg_u, MU);
  gru_scan<<<dim3(B_ / 16, 2), 256, SCAN_LDS, stream>>>(xg_u, Uc_u, bhf_u, bhb_u, u_out, B_, U_);

  // packed word-output scatter
  compute_offs<<<B_, 32, 0, stream>>>(u_len, w_len, offs, wl2);
  zero_f4<<<(B_ * U_ * WT_ * H_ / 4) / 256, 256, 0, stream>>>(w2, (size_t)B_ * U_ * WT_ * H_);
  scatter_w2<<<NSEQ, 256, 0, stream>>>(w_out, offs, wl2, w2);
}
